// Clahe2D_22067541967497
// MI455X (gfx1250) — matrix operand order repaired and checked by reading, not compiled, not run on hardware
//
#include <hip/hip_runtime.h>
#include <hip/hip_bf16.h>
#include <math.h>

// ---------------------------------------------------------------------------
// CLAHE (soft-KDE histogram + clip/redistribute + WMMA prefix-sum CDF +
// quintic grid-pull resample + global min/max normalize) for MI455X/gfx1250.
//
// Shapes fixed per reference: x = (B=2, C=1, H=512, W=512) fp32,
// GH=GW=8 tiles of 64x64, N_BINS=256, bandwidth 1e-3, clip limit 64.
// ---------------------------------------------------------------------------

typedef __attribute__((ext_vector_type(16))) _Float16 v16h;
typedef __attribute__((ext_vector_type(8)))  float    v8f;

#define NB       256
#define GHN      8
#define GWN      8
#define TILE_H   64
#define TILE_W   64
#define VOX      4096         // 64*64
#define IMG_H    512
#define IMG_W    512
#define IMG_PIX  (IMG_H*IMG_W)
#define NROWS    128          // B*C*GH*GW histograms
#define CLIP_LIM 64.0f        // max(4.0*4096//256, 1)

// ---------------------------------------------------------------------------
// Kernel 1: per-tile soft histogram + clip + redistribute.
// One block per tile (128 blocks, 256 threads = 8 wave32).
// Thread t exclusively owns bin t -> no atomics, deterministic.
// Gaussian support: |v*255 - bin| < 4.5 covers every fp32-nonzero term.
// ---------------------------------------------------------------------------
__global__ void clahe_hist_kernel(const float* __restrict__ x,
                                  float* __restrict__ histos,
                                  unsigned* __restrict__ mm)
{
    __shared__ float tile[VOX];
    __shared__ float red[NB];
    const int tid = threadIdx.x;
    const int r   = blockIdx.x;                 // histogram row, 0..127
    if (r == 0 && tid == 0) {                   // init global min/max keys
        mm[0] = 0xFFFFFFFFu;                    // min over monotone keys
        mm[1] = 0u;                             // max over monotone keys
    }
    const int bc = r >> 6;
    const int t  = r & 63;
    const int gh = t >> 3, gw = t & 7;
    const float* img = x + (size_t)bc * IMG_PIX;

    // Stage the 64x64 tile into LDS (coalesced).
    for (int i = tid; i < VOX; i += 256) {
        int row = i >> 6, col = i & 63;
        tile[i] = img[(gh * TILE_H + row) * IMG_W + gw * TILE_W + col];
    }
    __syncthreads();

    const float binf = (float)tid;
    float h = 0.0f;
    for (int i = 0; i < VOX; i += 4) {
        float4 v4 = reinterpret_cast<const float4*>(tile)[i >> 2]; // LDS broadcast
        #pragma unroll
        for (int q = 0; q < 4; ++q) {
            float v = (&v4.x)[q];
            float d = fmaf(v, 255.0f, -binf);          // (v - bin/255)*255
            if (fabsf(d) < 4.5f) {                     // fp32-exact window
                float tt = d * (1000.0f / 255.0f);     // (v-bin/255)/1e-3
                h += __expf(-0.5f * tt * tt);
            }
        }
    }

    // sum over bins (fixed-order tree -> deterministic)
    red[tid] = h; __syncthreads();
    #pragma unroll
    for (int s = 128; s > 0; s >>= 1) {
        if (tid < s) red[tid] += red[tid + s];
        __syncthreads();
    }
    const float sum = red[0];
    __syncthreads();

    // histos = pdf_norm * VOX = h * V / (sum + 1e-10*V); clip at 64
    float ht  = h * ((float)VOX / (sum + 1e-10f * (float)VOX));
    float hcl = fminf(ht, CLIP_LIM);

    red[tid] = hcl; __syncthreads();
    #pragma unroll
    for (int s = 128; s > 0; s >>= 1) {
        if (tid < s) red[tid] += red[tid + s];
        __syncthreads();
    }
    const float sumclip  = red[0];
    const float clipped  = (float)VOX - sumclip;
    const float residual = fmodf(clipped, 256.0f);
    const float redist   = (clipped - residual) * (1.0f / 256.0f);
    const float fin = hcl + redist + (((float)tid < residual) ? 1.0f : 0.0f);

    histos[(size_t)r * NB + tid] = fin;
}

// ---------------------------------------------------------------------------
// Kernel 2: CDF via triangular WMMA matmul.
// cdf[r, c+n] = running[r] + sum_{k<=n} H[r, c+k]
//   -> D(16x16) = A(16x32 f16: H block, K16..31 = 0) x B(32x16: [L16;0]) + C
// with the running per-row totals carried in the f32 accumulator, updated
// each block from the N=15 column via a lane-15/31 broadcast.
// One wave per 16-histogram slab; 8 blocks x 32 threads; 16 WMMAs per slab.
// Output scattered into the (bc, bin, gh, gw) volume consumed by kernel 3.
// ---------------------------------------------------------------------------
__global__ void clahe_cdf_wmma_kernel(const float* __restrict__ histos,
                                      float* __restrict__ cdfvol)
{
    const int lane = threadIdx.x;      // 0..31 (wave32)
    const int half = lane >> 4;        // 0: lanes 0-15, 1: lanes 16-31
    const int lq   = lane & 15;
    const int slab = blockIdx.x;       // 0..7 -> rows slab*16 .. slab*16+15
    const int rowA = slab * 16 + lq;   // M index == lq in both halves

    // B = [L16 ; 0]:  B[k, n] = (k <= n), rows 16..31 zero.
    // Layout: lanes 0-15 elems 0..7 hold K=0..7, lanes 16-31 elems 0..7 hold K=8..15 (column N=lane%16); elems 8..15 hold K16+ = 0.
    v16h bm;
    #pragma unroll
    for (int e = 0; e < 16; ++e) {
        float bv = (e < 8 && half * 8 + e <= lq) ? 1.0f : 0.0f;
        bm[e] = (_Float16)bv;
    }

    v8f run = {0.f, 0.f, 0.f, 0.f, 0.f, 0.f, 0.f, 0.f};   // running row sums

    for (int j = 0; j < 16; ++j) {
        const int c = j * 16;
        // A layout (16-bit 16x32): lane<16 -> elems 0..7 = K0..7 (cols c..c+7),
        // lane>=16 -> elems 0..7 = K8..15 (cols c+8..c+15); elems 8..15 = K16+ = 0.
        const float* hp = histos + (size_t)rowA * NB + c + half * 8;
        float4 f0 = reinterpret_cast<const float4*>(hp)[0];
        float4 f1 = reinterpret_cast<const float4*>(hp)[1];
        v16h am;
        am[0] = (_Float16)f0.x; am[1] = (_Float16)f0.y;
        am[2] = (_Float16)f0.z; am[3] = (_Float16)f0.w;
        am[4] = (_Float16)f1.x; am[5] = (_Float16)f1.y;
        am[6] = (_Float16)f1.z; am[7] = (_Float16)f1.w;
        #pragma unroll
        for (int e = 8; e < 16; ++e) am[e] = (_Float16)0.0f;

        v8f d = __builtin_amdgcn_wmma_f32_16x16x32_f16(
            /*neg_a=*/false, am, /*neg_b=*/false, bm,
            /*c_mod=*/(short)0, run, /*reuse_a=*/false, /*reuse_b=*/false);

        // New running total per row = value at N=15 (lane 15 for M=0..7,
        // lane 31 for M=8..15 of each accumulator VGPR).
        const int src = (half == 0) ? 15 : 31;
        #pragma unroll
        for (int p = 0; p < 8; ++p) {
            run[p] = __shfl(d[p], src, 32);
            const int M  = p + half * 8;
            const int r  = slab * 16 + M;
            const int bc = r >> 6, tileIdx = r & 63;
            float v = d[p] * (255.0f / 4096.0f);
            v = fminf(fmaxf(v, 0.0f), 255.0f);
            cdfvol[(size_t)bc * (NB * 64) + (c + lq) * 64 + tileIdx] = v;
        }
    }
}

// ---------------------------------------------------------------------------
// Kernel 3: quintic grid-pull resample from the (256,8,8) CDF LUT in LDS.
// 64 KB LUT per image fits easily in CDNA5's 320 KB/WGP LDS.
// 128 blocks (one per 64x64 output patch), 256 threads, 16 px/thread.
// ---------------------------------------------------------------------------
__device__ __forceinline__ float bspline5(float t)
{
    float x  = fabsf(t);
    float x2 = x * x, x4 = x2 * x2;
    if (x < 1.0f)
        return 0.55f - 0.5f * x2 + 0.25f * x4 - x4 * x * (1.0f / 12.0f);
    if (x < 2.0f)
        return 0.425f + 0.625f * x - 1.75f * x2 + 1.25f * x2 * x
               - 0.375f * x4 + x4 * x * (1.0f / 24.0f);
    if (x < 3.0f) {
        float y = 3.0f - x, y2 = y * y;
        return y2 * y2 * y * (1.0f / 120.0f);
    }
    return 0.0f;
}

__device__ __forceinline__ int reflect_idx(int i, int n)
{
    int p = 2 * n;
    int m = i % p; if (m < 0) m += p;
    return (m >= n) ? (p - 1 - m) : m;
}

__global__ void clahe_resample_kernel(const float* __restrict__ x,
                                      const float* __restrict__ cdfvol,
                                      float* __restrict__ out,
                                      unsigned* __restrict__ mm)
{
    extern __shared__ float lut[];              // NB*64 floats = 64 KB
    const int tid = threadIdx.x;
    const int b   = blockIdx.x;                 // 0..127
    const int bc  = b >> 6, t = b & 63;
    const int tr  = (t >> 3) * 64, tc = (t & 7) * 64;

    const float* cv = cdfvol + (size_t)bc * (NB * 64);
    for (int i = tid; i < NB * 64; i += 256) lut[i] = cv[i];
    __syncthreads();

    const float* img = x + (size_t)bc * IMG_PIX;
    float* op = out + (size_t)bc * IMG_PIX;

    const float lo    = -0.53125f;              // -0.5 - 0.25/8
    const float stepc = 8.0625f / 511.0f;       // linspace step (H==W==512)

    float vmin =  3.0e38f, vmax = -3.0e38f;

    for (int it = 0; it < 16; ++it) {
        const int pix  = tid + 256 * it;
        const int prow = pix >> 6, pcol = pix & 63;
        const int hh = tr + prow, ww = tc + pcol;
        const float v = img[hh * IMG_W + ww];
        if (it + 1 < 16) {                       // gfx1250 global_prefetch_b8
            const int np = pix + 256;
            __builtin_prefetch(&img[(tr + (np >> 6)) * IMG_W + tc + (np & 63)], 0, 1);
        }
        const float binc = v * 255.0f;
        const float chc  = fmaf((float)hh, stepc, lo);
        const float cwc  = fmaf((float)ww, stepc, lo);

        const int b0 = (int)floorf(binc);
        const int h0 = (int)floorf(chc);
        const int w0 = (int)floorf(cwc);

        float wb[6], whv[6], wwv[6];
        int   ibv[6], ihv[6], iwv[6];
        #pragma unroll
        for (int o = 0; o < 6; ++o) {
            int nb = b0 + o - 2;
            wb [o] = bspline5(binc - (float)nb); ibv[o] = reflect_idx(nb, NB);
            int nh = h0 + o - 2;
            whv[o] = bspline5(chc - (float)nh);  ihv[o] = reflect_idx(nh, GHN);
            int nw = w0 + o - 2;
            wwv[o] = bspline5(cwc - (float)nw);  iwv[o] = reflect_idx(nw, GWN);
        }

        float acc = 0.0f;
        for (int i = 0; i < 6; ++i) {
            const float* plane = lut + ibv[i] * 64;
            float accj = 0.0f;
            #pragma unroll
            for (int jj = 0; jj < 6; ++jj) {
                const float* rowp = plane + ihv[jj] * 8;
                float s = 0.0f;
                #pragma unroll
                for (int k = 0; k < 6; ++k) s = fmaf(rowp[iwv[k]], wwv[k], s);
                accj = fmaf(whv[jj], s, accj);
            }
            acc = fmaf(wb[i], accj, acc);
        }
        op[hh * IMG_W + ww] = acc;
        vmin = fminf(vmin, acc);
        vmax = fmaxf(vmax, acc);
    }

    // wave32 min/max reduce, then one atomic per wave (order-independent).
    #pragma unroll
    for (int off = 16; off > 0; off >>= 1) {
        vmin = fminf(vmin, __shfl_down(vmin, off, 32));
        vmax = fmaxf(vmax, __shfl_down(vmax, off, 32));
    }
    if ((tid & 31) == 0) {
        unsigned kb = __float_as_uint(vmin);
        unsigned kmin = (kb & 0x80000000u) ? ~kb : (kb ^ 0x80000000u);
        kb = __float_as_uint(vmax);
        unsigned kmax = (kb & 0x80000000u) ? ~kb : (kb ^ 0x80000000u);
        atomicMin(&mm[0], kmin);
        atomicMax(&mm[1], kmax);
    }
}

// ---------------------------------------------------------------------------
// Kernel 4: global normalize (out - min) / (max - min + 1e-10)
// ---------------------------------------------------------------------------
__device__ __forceinline__ float mm_decode(unsigned u)
{
    u = (u & 0x80000000u) ? (u ^ 0x80000000u) : ~u;
    return __uint_as_float(u);
}

__global__ void clahe_norm_kernel(float* __restrict__ out,
                                  const unsigned* __restrict__ mm, int n)
{
    const int i = blockIdx.x * 256 + threadIdx.x;
    if (i >= n) return;
    const float mn = mm_decode(mm[0]);
    const float mx = mm_decode(mm[1]);
    out[i] = (out[i] - mn) / (mx - mn + 1e-10f);
}

// ---------------------------------------------------------------------------
extern "C" void kernel_launch(void* const* d_in, const int* in_sizes, int n_in,
                              void* d_out, int out_size, void* d_ws, size_t ws_size,
                              hipStream_t stream)
{
    const float* x   = (const float*)d_in[0];
    float*       out = (float*)d_out;

    // Workspace: histos (128*256 f32) | cdfvol (128*256 f32) | minmax (2 u32)
    float*    histos = (float*)d_ws;
    float*    cdfvol = histos + (size_t)NROWS * NB;
    unsigned* mm     = (unsigned*)(cdfvol + (size_t)NROWS * NB);

    clahe_hist_kernel   <<<NROWS, 256, 0, stream>>>(x, histos, mm);
    clahe_cdf_wmma_kernel<<<8, 32, 0, stream>>>(histos, cdfvol);
    clahe_resample_kernel<<<128, 256, NB * 64 * sizeof(float), stream>>>(x, cdfvol, out, mm);
    clahe_norm_kernel   <<<(out_size + 255) / 256, 256, 0, stream>>>(out, mm, out_size);
}
